// MultiHeadAttention_62878321213626
// MI455X (gfx1250) — matrix operand order repaired and checked by reading, not compiled, not run on hardware
//
#include <hip/hip_runtime.h>
#include <math.h>

#define B_   2
#define S_   2048
#define HID_ 768
#define NH_  12
#define HD_  64

typedef __attribute__((ext_vector_type(16))) __bf16        v16bf;
typedef __attribute__((ext_vector_type(8)))  float         v8f;
typedef __attribute__((ext_vector_type(8)))  unsigned int  v8u;

union FragU { v8u u; uint4 q[2]; };

static __device__ inline v16bf as_bf(const FragU& f) {
    return __builtin_bit_cast(v16bf, f.u);
}

static __device__ inline v8f wmma_bf16(v16bf a, v16bf b, v8f c) {
    return __builtin_amdgcn_wmma_f32_16x16x32_bf16(
               false, a, false, b, (short)0, c, false, false);
}

static __device__ inline unsigned short f2bf(float f) {
    unsigned int u = __builtin_bit_cast(unsigned int, f);
    u += 0x7FFFu + ((u >> 16) & 1u);          // round-to-nearest-even
    return (unsigned short)(u >> 16);
}

// async copy of 16B global -> LDS, tracked by ASYNCcnt
static __device__ inline void async_b128(unsigned lds_addr, const void* gaddr) {
    asm volatile("global_load_async_to_lds_b128 %0, %1, off"
                 :: "v"(lds_addr), "v"(gaddr) : "memory");
}
static __device__ inline unsigned lds_off(const void* p) {
    return (unsigned)(uintptr_t)p;            // generic LDS ptr: low 32 bits = LDS offset
}

// ---------------------------------------------------------------- fp32 -> bf16
__global__ void cvt_bf16(const float* __restrict__ src,
                         unsigned short* __restrict__ dst, int n) {
    int i = blockIdx.x * blockDim.x + threadIdx.x;
    if (i < n) dst[i] = f2bf(src[i]);
}

// -------------------------------------------------- C[M][N] = A[M][K] * Bt[N][K]^T
// 32x32 tile per wave (2x2 register blocking), 4 waves -> 64x64 per block.
__global__ __launch_bounds__(128) void gemm_bf16(
    const unsigned short* __restrict__ A,   // [M][K] row-major bf16
    const unsigned short* __restrict__ Bt,  // [N][K] row-major bf16
    float* __restrict__ C, int M, int N, int K)
{
    const int lane = threadIdx.x & 31;
    const int wave = threadIdx.x >> 5;
    const int mw   = wave >> 1, nw = wave & 1;        // 2x2 wave grid
    const int nGrp = N >> 6;
    const int mblk = blockIdx.x / nGrp;
    const int nblk = blockIdx.x % nGrp;
    const int n    = lane & 15;
    const int half = lane >> 4;
    const int aoff = half ? 8 : 0;       // A 16-bit layout: lanes 16-31 hold K+8
    const int boff = half * 8;           // B 16-bit layout: lanes 16-31 hold K+8

    const unsigned short* Ap0 = A  + (size_t)(mblk * 64 + mw * 32 + n) * K;
    const unsigned short* Ap1 = Ap0 + (size_t)16 * K;
    const unsigned short* Bp0 = Bt + (size_t)(nblk * 64 + nw * 32 + n) * K;
    const unsigned short* Bp1 = Bp0 + (size_t)16 * K;

    v8f acc[2][2] = {};
    for (int kc = 0; kc < K; kc += 32) {
        FragU fa0, fa1, fb0, fb1;
        fa0.q[0] = *(const uint4*)(Ap0 + kc + aoff);
        fa0.q[1] = *(const uint4*)(Ap0 + kc + aoff + 16);
        fa1.q[0] = *(const uint4*)(Ap1 + kc + aoff);
        fa1.q[1] = *(const uint4*)(Ap1 + kc + aoff + 16);
        fb0.q[0] = *(const uint4*)(Bp0 + kc + boff);
        fb0.q[1] = *(const uint4*)(Bp0 + kc + boff + 16);
        fb1.q[0] = *(const uint4*)(Bp1 + kc + boff);
        fb1.q[1] = *(const uint4*)(Bp1 + kc + boff + 16);
        v16bf a0 = as_bf(fa0), a1 = as_bf(fa1);
        v16bf b0 = as_bf(fb0), b1 = as_bf(fb1);
        acc[0][0] = wmma_bf16(a0, b0, acc[0][0]);
        acc[0][1] = wmma_bf16(a0, b1, acc[0][1]);
        acc[1][0] = wmma_bf16(a1, b0, acc[1][0]);
        acc[1][1] = wmma_bf16(a1, b1, acc[1][1]);
    }
    // C layout: VGPR r -> row r + 8*half, col = lane&15
    #pragma unroll
    for (int mt = 0; mt < 2; ++mt)
        #pragma unroll
        for (int nt = 0; nt < 2; ++nt) {
            int rowb = mblk * 64 + mw * 32 + mt * 16 + half * 8;
            int colb = nblk * 64 + nw * 32 + nt * 16 + n;
            #pragma unroll
            for (int r = 0; r < 8; ++r)
                C[(size_t)(rowb + r) * N + colb] = acc[mt][nt][r];
        }
}

// --------------------------------------- RoPE (head-indexed per reference) + scatter
__global__ void rope_scatter(const float* __restrict__ qkv,       // [B*S][2304]
                             unsigned short* __restrict__ Qb,     // [B,NH,S,HD]
                             unsigned short* __restrict__ Kb,     // [B,NH,S,HD]
                             unsigned short* __restrict__ Vt)     // [B,NH,HD,S]
{
    int i = blockIdx.x * blockDim.x + threadIdx.x;
    if (i >= B_ * S_ * NH_ * HD_) return;
    int d = i & (HD_ - 1);
    int h = (i / HD_) % NH_;
    int s = (i / (HD_ * NH_)) % S_;
    int b = i / (HD_ * NH_ * S_);

    const float* base = qkv + ((size_t)(b * S_ + s)) * (3 * HID_) + h * 3 * HD_;
    float q = base[d], k = base[HD_ + d], v = base[2 * HD_ + d];

    int j = d & 31;                                   // inv_freq index
    // angle = h * 10000^(-j/32)   (reference uses head index, not position)
    float ang = (float)h * exp2f(-(float)j * (13.287712379549449f / 32.0f));
    float c  = cosf(ang), sn = sinf(ang);
    float qr = (d < 32) ? -base[d + 32]        : base[d - 32];
    float kr = (d < 32) ? -base[HD_ + d + 32]  : base[HD_ + d - 32];
    float qo = q * c + qr * sn;
    float ko = k * c + kr * sn;

    size_t bh = (size_t)b * NH_ + h;
    Qb[(bh * S_ + s) * HD_ + d] = f2bf(qo);
    Kb[(bh * S_ + s) * HD_ + d] = f2bf(ko);
    Vt[(bh * HD_ + d) * S_ + s] = f2bf(v);
}

// ----------------------------------------------------------- flash attention
// 4 waves/block, same (b,h), consecutive q-tiles. K/V tiles double-buffered in
// LDS via global_load_async_to_lds_b128 (ASYNCcnt); online softmax in fp32.
__global__ __launch_bounds__(128) void flash_attn(
    const unsigned short* __restrict__ Qb,
    const unsigned short* __restrict__ Kb,
    const unsigned short* __restrict__ Vt,
    const float* __restrict__ mask,          // [B][S]
    unsigned short* __restrict__ ctxb)       // [B*S][HID] bf16
{
    __shared__ __align__(16) unsigned short Klds[2][32 * HD_]; // 2 x 4KB: [key][d]
    __shared__ __align__(16) unsigned short Vlds[2][HD_ * 32]; // 2 x 4KB: [d][key]
    __shared__ __align__(16) float          Plds[4][16 * 32];  // per-wave P staging

    const int tid  = threadIdx.x;
    const int lane = tid & 31;
    const int wave = tid >> 5;
    float* P = &Plds[wave][0];

    const int tile = blockIdx.x * 4 + wave;
    const int nQT  = S_ / 16;
    const int qt   = tile % nQT;
    const int h    = (tile / nQT) % NH_;      // uniform across the block
    const int b    = tile / (nQT * NH_);      // uniform across the block
    const size_t bh = (size_t)b * NH_ + h;
    const int n    = lane & 15;
    const int half = lane >> 4;

    const unsigned short* Kbase = Kb + bh * (size_t)S_ * HD_;
    const unsigned short* Vbase = Vt + bh * (size_t)HD_ * S_;
    const float* mrow = mask + (size_t)b * S_;

    // Q A-fragments: two K-chunks of 32 over HD=64
    v16bf qa[2];
    {
        const unsigned short* Qp = Qb + (bh * S_ + qt * 16 + n) * HD_;
        #pragma unroll
        for (int c2 = 0; c2 < 2; ++c2) {
            FragU f;
            int koff = c2 * 32 + (half ? 8 : 0);
            f.q[0] = *(const uint4*)(Qp + koff);
            f.q[1] = *(const uint4*)(Qp + koff + 16);
            qa[c2] = as_bf(f);
        }
    }

    float m_i[8], l_i[8];
    #pragma unroll
    for (int r = 0; r < 8; ++r) { m_i[r] = -INFINITY; l_i[r] = 0.0f; }
    v8f O[4] = {};

    // ---- async stage of one 32-key K/V tile (block-cooperative, 128 threads)
    auto stage = [&](int buf, int jb) {
        // K tile: contiguous 4KB (32 keys x 64 d x bf16)
        const char* ksrc = (const char*)(Kbase + (size_t)jb * HD_);
        unsigned kdst = lds_off(&Klds[buf][0]);
        #pragma unroll
        for (int rep = 0; rep < 2; ++rep) {
            int c = tid + rep * 128;                      // 256 x 16B chunks
            async_b128(kdst + c * 16, ksrc + c * 16);
        }
        // V tile: 64 rows x 64B, global row stride S_ elements
        unsigned vdst = lds_off(&Vlds[buf][0]);
        #pragma unroll
        for (int rep = 0; rep < 2; ++rep) {
            int c   = tid + rep * 128;                    // 256 chunks
            int row = c >> 2, seg = c & 3;
            const char* g = (const char*)(Vbase + (size_t)row * S_ + jb) + seg * 16;
            async_b128(vdst + (row * 32 + seg * 8) * 2, g);
        }
    };

    stage(0, 0);

    for (int jb = 0; jb < S_; jb += 32) {
        const int buf = (jb >> 5) & 1;
        asm volatile("s_wait_asynccnt 0x0" ::: "memory");
        __syncthreads();                                   // tile visible to all waves
        if (jb + 32 < S_) stage(buf ^ 1, jb + 32);         // prefetch next tile

        // ---- scores: two 16x16 tiles over 32 keys, K-dim = HD (2 chunks)
        v8f s0 = {}, s1 = {};
        #pragma unroll
        for (int c2 = 0; c2 < 2; ++c2) {
            int koff = c2 * 32 + half * 8;
            const unsigned short* K0 = &Klds[buf][(n)      * HD_ + koff];
            const unsigned short* K1 = &Klds[buf][(16 + n) * HD_ + koff];
            FragU f0, f1;
            f0.q[0] = *(const uint4*)(K0); f0.q[1] = *(const uint4*)(K0 + 16);
            f1.q[0] = *(const uint4*)(K1); f1.q[1] = *(const uint4*)(K1 + 16);
            s0 = wmma_bf16(qa[c2], as_bf(f0), s0);
            s1 = wmma_bf16(qa[c2], as_bf(f1), s1);
        }
        float bias0 = (1.0f - mrow[jb + n])      * -10000.0f;
        float bias1 = (1.0f - mrow[jb + 16 + n]) * -10000.0f;

        // ---- online softmax (row r lives in VGPR r; halves = rows r / r+8)
        float alpha[8], p0[8], p1[8];
        #pragma unroll
        for (int r = 0; r < 8; ++r) {
            float v0 = s0[r] * 0.125f + bias0;
            float v1 = s1[r] * 0.125f + bias1;
            float mx = fmaxf(v0, v1);
            #pragma unroll
            for (int d2 = 1; d2 < 16; d2 <<= 1)
                mx = fmaxf(mx, __shfl_xor(mx, d2, 32));
            float mn = fmaxf(m_i[r], mx);
            alpha[r] = __expf(m_i[r] - mn);
            m_i[r] = mn;
            p0[r] = __expf(v0 - mn);
            p1[r] = __expf(v1 - mn);
            float rs = p0[r] + p1[r];
            #pragma unroll
            for (int d2 = 1; d2 < 16; d2 <<= 1)
                rs += __shfl_xor(rs, d2, 32);
            l_i[r] = l_i[r] * alpha[r] + rs;
        }
        #pragma unroll
        for (int t = 0; t < 4; ++t)
            #pragma unroll
            for (int r = 0; r < 8; ++r)
                O[t][r] *= alpha[r];

        // ---- C-layout fp32 -> LDS -> A-layout bf16 (cross-lane, needs DS wait)
        #pragma unroll
        for (int r = 0; r < 8; ++r) {
            int row = r + half * 8;
            P[row * 32 + n]      = p0[r];
            P[row * 32 + 16 + n] = p1[r];
        }
        asm volatile("s_wait_dscnt 0x0" ::: "memory");

        FragU pf;
        const int kb2 = half ? 8 : 0;
        #pragma unroll
        for (int e = 0; e < 4; ++e) {
            unsigned lo  = f2bf(P[n * 32 + kb2 + 2 * e]);
            unsigned hi  = f2bf(P[n * 32 + kb2 + 2 * e + 1]);
            pf.u[e]      = lo | (hi << 16);
            unsigned lo2 = f2bf(P[n * 32 + kb2 + 16 + 2 * e]);
            unsigned hi2 = f2bf(P[n * 32 + kb2 + 16 + 2 * e + 1]);
            pf.u[4 + e]  = lo2 | (hi2 << 16);
        }

        // ---- O += P(16x32) * V(32x64): 4 N-tiles of 16
        #pragma unroll
        for (int t = 0; t < 4; ++t) {
            const unsigned short* Vp = &Vlds[buf][(t * 16 + n) * 32 + half * 8];
            FragU fv;
            fv.q[0] = *(const uint4*)(Vp);
            fv.q[1] = *(const uint4*)(Vp + 16);
            O[t] = wmma_bf16(as_bf(pf), as_bf(fv), O[t]);
        }

        asm volatile("s_wait_dscnt 0x0" ::: "memory");     // all LDS reads retired
        __syncthreads();                                   // before buffer reuse
    }

    // ---- epilogue: normalize + store ctx in [b,s, h*64+d] bf16
    #pragma unroll
    for (int t = 0; t < 4; ++t) {
        #pragma unroll
        for (int r = 0; r < 8; ++r) {
            float o = O[t][r] / l_i[r];
            int row = qt * 16 + r + half * 8;
            ctxb[((size_t)b * S_ + row) * HID_ + h * HD_ + t * 16 + n] = f2bf(o);
        }
    }
}

// ---------------------------------------------------------------------------
extern "C" void kernel_launch(void* const* d_in, const int* in_sizes, int n_in,
                              void* d_out, int out_size, void* d_ws, size_t ws_size,
                              hipStream_t stream)
{
    const float* x    = (const float*)d_in[0];
    const float* mask = (const float*)d_in[1];
    const float* wqkv = (const float*)d_in[2];
    const float* wout = (const float*)d_in[3];
    float* out = (float*)d_out;

    char* ws = (char*)d_ws;
    size_t off = 0;
    auto alloc = [&](size_t bytes) -> char* {
        char* p = ws + off;
        off += (bytes + 255) & ~(size_t)255;
        return p;
    };
    unsigned short* xb    = (unsigned short*)alloc((size_t)B_ * S_ * HID_ * 2);
    unsigned short* wqkvb = (unsigned short*)alloc((size_t)3 * HID_ * HID_ * 2);
    unsigned short* woutb = (unsigned short*)alloc((size_t)HID_ * HID_ * 2);
    float*          qkv   = (float*)alloc((size_t)B_ * S_ * 3 * HID_ * 4);
    unsigned short* Qb    = (unsigned short*)alloc((size_t)B_ * NH_ * S_ * HD_ * 2);
    unsigned short* Kb    = (unsigned short*)alloc((size_t)B_ * NH_ * S_ * HD_ * 2);
    unsigned short* Vt    = (unsigned short*)alloc((size_t)B_ * NH_ * S_ * HD_ * 2);
    unsigned short* ctxb  = (unsigned short*)alloc((size_t)B_ * S_ * HID_ * 2);

    int n1 = B_ * S_ * HID_;
    cvt_bf16<<<(n1 + 255) / 256, 256, 0, stream>>>(x, xb, n1);
    int n2 = 3 * HID_ * HID_;
    cvt_bf16<<<(n2 + 255) / 256, 256, 0, stream>>>(wqkv, wqkvb, n2);
    int n3 = HID_ * HID_;
    cvt_bf16<<<(n3 + 255) / 256, 256, 0, stream>>>(wout, woutb, n3);

    {   // QKV projection: [4096 x 2304]
        int M = B_ * S_, N = 3 * HID_, K = HID_;
        gemm_bf16<<<(M / 64) * (N / 64), 128, 0, stream>>>(xb, wqkvb, qkv, M, N, K);
    }
    {   // RoPE + scatter into attention layouts
        int n = B_ * S_ * NH_ * HD_;
        rope_scatter<<<(n + 255) / 256, 256, 0, stream>>>(qkv, Qb, Kb, Vt);
    }
    {   // flash attention: 3072 query tiles, 4 waves/block
        int tiles = B_ * NH_ * (S_ / 16);
        flash_attn<<<tiles / 4, 128, 0, stream>>>(Qb, Kb, Vt, mask, ctxb);
    }
    {   // output projection: [4096 x 768] -> fp32 d_out
        int M = B_ * S_, N = HID_, K = HID_;
        gemm_bf16<<<(M / 64) * (N / 64), 128, 0, stream>>>(ctxb, woutb, out, M, N, K);
    }
}
